// LinearizedMMDLoss_78297253806512
// MI455X (gfx1250) — matrix operand order repaired and checked by reading, not compiled, not run on hardware
//
#include <hip/hip_runtime.h>

typedef __attribute__((ext_vector_type(16))) _Float16 v16h;
typedef __attribute__((ext_vector_type(8)))  float    v8f;
typedef __attribute__((ext_vector_type(4)))  float    v4f;

#define N_FEAT      32
#define N_FOURIER   512
#define N_KERNELS   5
#define N_ROWS      16384
#define MTILES_PER_JOB 16                         /* 256 rows per wave-job   */
#define NTILES      (N_FOURIER / 16)              /* 32                      */
#define MCHUNKS     (N_ROWS / (16 * MTILES_PER_JOB)) /* 64                   */
#define JOBS        (2 * NTILES * MCHUNKS)        /* 4096 wave-jobs          */
#define WAVES_PER_BLOCK 8
#define BLOCKS      (JOBS / WAVES_PER_BLOCK)      /* 512                     */
#define BUF_ELEMS   (2 * N_KERNELS * 2 * N_FOURIER) /* 10240 floats          */

__device__ __forceinline__ void split_f16(float x, _Float16& hi, _Float16& lo) {
  hi = (_Float16)x;
  lo = (_Float16)(x - (float)hi);   // residual: recovers ~f32 GEMM accuracy
}

__global__ void zero_ws_kernel(float* __restrict__ p, int n) {
  int i = blockIdx.x * blockDim.x + threadIdx.x;
  if (i < n) p[i] = 0.0f;
}

// Fused: Z @ omegas^T (split-f16 WMMA) -> per-sigma sincos -> batch-sum via atomics.
// buf layout: buf[((z*N_KERNELS + k)*2 + cs)*N_FOURIER + r], cs: 0=cos 1=sin
__global__ __launch_bounds__(256) void mmd_proj_kernel(
    const float* __restrict__ X, const float* __restrict__ Y,
    const float* __restrict__ omegas, const float* __restrict__ mult,
    float* __restrict__ buf)
{
  const int lane = threadIdx.x & 31;
  const int wave = threadIdx.x >> 5;
  const int job  = blockIdx.x * WAVES_PER_BLOCK + wave;

  const int z      = job >> 11;      // 2048 jobs per matrix
  const int rem    = job & 2047;
  const int ntile  = rem >> 6;       // 0..31  (fourier tile)
  const int mchunk = rem & 63;       // 0..63  (row chunk)

  const float* __restrict__ Z = z ? Y : X;

  // inverse sigmas; sigma0 = sqrt(SCALE) = 1
  float invsig[N_KERNELS];
#pragma unroll
  for (int k = 0; k < N_KERNELS; ++k) invsig[k] = 1.0f / mult[k];

  // ---- B fragment (omegas): 32x16 (KxN), contiguous K order, matching A ----
  // lane L: column N = L%16 -> fourier row r; halves kh = L/16 select K=kh*16..kh*16+15
  const int mrow = lane & 15;
  const int kh   = lane >> 4;
  const int r    = ntile * 16 + mrow;

  const v4f* bp = (const v4f*)(omegas + r * N_FEAT + kh * 16);
  v4f bq0 = bp[0], bq1 = bp[1], bq2 = bp[2], bq3 = bp[3];
  v16h b_hi, b_lo;
#pragma unroll
  for (int j = 0; j < 4; ++j) {
    _Float16 h, l;
    split_f16(bq0[j], h, l); b_hi[j]      = h; b_lo[j]      = l;
    split_f16(bq1[j], h, l); b_hi[4 + j]  = h; b_lo[4 + j]  = l;
    split_f16(bq2[j], h, l); b_hi[8 + j]  = h; b_lo[8 + j]  = l;
    split_f16(bq3[j], h, l); b_hi[12 + j] = h; b_lo[12 + j] = l;
  }

  float scos[N_KERNELS], ssin[N_KERNELS];
#pragma unroll
  for (int k = 0; k < N_KERNELS; ++k) { scos[k] = 0.0f; ssin[k] = 0.0f; }

  // A-element K offsets (contiguous, matching B): elems 0..7 -> K=kh*16.., elems 8..15 -> K=kh*16+8..
  const int koff0 = kh * 16;
  const int koff1 = kh * 16 + 8;

  for (int t = 0; t < MTILES_PER_JOB; ++t) {
    const int m = (mchunk * MTILES_PER_JOB + t) * 16 + mrow;
    const float* rowp = Z + m * N_FEAT;
    const v4f* a0 = (const v4f*)(rowp + koff0);
    const v4f* a1 = (const v4f*)(rowp + koff1);
    v4f aq0 = a0[0], aq1 = a0[1], aq2 = a1[0], aq3 = a1[1];

    v16h a_hi, a_lo;
#pragma unroll
    for (int j = 0; j < 4; ++j) {
      _Float16 h, l;
      split_f16(aq0[j], h, l); a_hi[j]      = h; a_lo[j]      = l;
      split_f16(aq1[j], h, l); a_hi[4 + j]  = h; a_lo[4 + j]  = l;
      split_f16(aq2[j], h, l); a_hi[8 + j]  = h; a_lo[8 + j]  = l;
      split_f16(aq3[j], h, l); a_hi[12 + j] = h; a_lo[12 + j] = l;
    }

    // base tile = A*B in ~f32 precision via 3 f16 WMMAs
    v8f c = {};
    c = __builtin_amdgcn_wmma_f32_16x16x32_f16(false, a_hi, false, b_hi,
                                               (short)0, c, false, false);
    c = __builtin_amdgcn_wmma_f32_16x16x32_f16(false, a_lo, false, b_hi,
                                               (short)0, c, false, false);
    c = __builtin_amdgcn_wmma_f32_16x16x32_f16(false, a_hi, false, b_lo,
                                               (short)0, c, false, false);

    // per-lane: 8 rows of base for column r; accumulate cos/sin over the batch axis
#pragma unroll
    for (int v = 0; v < 8; ++v) {
      const float basev = c[v];
#pragma unroll
      for (int k = 0; k < N_KERNELS; ++k) {
        const float arg = basev * invsig[k];
        scos[k] += __cosf(arg);
        ssin[k] += __sinf(arg);
      }
    }
  }

  // lanes L and L+16 hold the same fourier column r -> atomics merge them
#pragma unroll
  for (int k = 0; k < N_KERNELS; ++k) {
    atomicAdd(&buf[(((z * N_KERNELS) + k) * 2 + 0) * N_FOURIER + r], scos[k]);
    atomicAdd(&buf[(((z * N_KERNELS) + k) * 2 + 1) * N_FOURIER + r], ssin[k]);
  }
}

__global__ __launch_bounds__(256) void finalize_kernel(
    const float* __restrict__ buf, float* __restrict__ out,
    float inv_ns, float inv_nt)
{
  __shared__ float red[256];
  const int tid = threadIdx.x;
  float acc = 0.0f;
  // first half of buf = X sums, second half = Y sums, identical (k,cs,r) order
  for (int i = tid; i < N_KERNELS * 2 * N_FOURIER; i += 256) {
    const float d = buf[i] * inv_ns - buf[N_KERNELS * 2 * N_FOURIER + i] * inv_nt;
    acc += d * d;
  }
  red[tid] = acc;
  __syncthreads();
  for (int s = 128; s > 0; s >>= 1) {
    if (tid < s) red[tid] += red[tid + s];
    __syncthreads();
  }
  if (tid == 0) out[0] = red[0] * (2.0f / (float)N_FOURIER) / (float)N_KERNELS;
}

extern "C" void kernel_launch(void* const* d_in, const int* in_sizes, int n_in,
                              void* d_out, int out_size, void* d_ws, size_t ws_size,
                              hipStream_t stream) {
  const float* X      = (const float*)d_in[0];
  const float* Y      = (const float*)d_in[1];
  const float* omegas = (const float*)d_in[2];
  const float* mult   = (const float*)d_in[3];
  float* out = (float*)d_out;
  float* buf = (float*)d_ws;  // 10240 floats = 40 KB

  hipLaunchKernelGGL(zero_ws_kernel, dim3((BUF_ELEMS + 255) / 256), dim3(256),
                     0, stream, buf, BUF_ELEMS);

  hipLaunchKernelGGL(mmd_proj_kernel, dim3(BLOCKS), dim3(256), 0, stream,
                     X, Y, omegas, mult, buf);

  const float inv_ns = 1.0f / (float)(in_sizes[0] / N_FEAT);
  const float inv_nt = 1.0f / (float)(in_sizes[1] / N_FEAT);
  hipLaunchKernelGGL(finalize_kernel, dim3(1), dim3(256), 0, stream,
                     buf, out, inv_ns, inv_nt);
}
